// AudioUNet5ConvLSTM_37194416783461
// MI455X (gfx1250) — matrix operand order repaired and checked by reading, not compiled, not run on hardware
//
#include <hip/hip_runtime.h>
#include <hip/hip_fp16.h>

typedef __attribute__((ext_vector_type(16))) _Float16 v16h;
typedef __attribute__((ext_vector_type(8)))  float    v8f;
typedef __attribute__((ext_vector_type(4)))  float    v4f;
typedef int b128v __attribute__((vector_size(16)));   // matches async-builtin param type

#define TQ 512   // time dimension (constant for every layer)
#define BQ 16    // batch

#if defined(__has_builtin)
#  if __has_builtin(__builtin_amdgcn_global_load_async_to_lds_b128)
#    define HAVE_ASYNC_LDS 1
#  endif
#endif

union HalfVec {
  v16h v;
  v4f  f4[2];
};
union Half8 {
  v4f f;
  _Float16 h[8];
};

// 16B-aligned LDS load of a 16-half B fragment slice (two ds_load_b128)
__device__ __forceinline__ v16h lds_v16h(const _Float16* p) {
  HalfVec u;
  u.f4[0] = *(const v4f*)(p);
  u.f4[1] = *(const v4f*)(p + 8);
  return u.v;
}

// 16-byte global(f16) -> LDS copy: async DMA on gfx1250 when available
__device__ __forceinline__ void a_copy16(const _Float16* g, _Float16* l) {
#ifdef HAVE_ASYNC_LDS
  __builtin_amdgcn_global_load_async_to_lds_b128(
      (__attribute__((address_space(1))) b128v*)g,
      (__attribute__((address_space(3))) b128v*)l, 0, 0);
#else
  *(v4f*)l = *(const v4f*)g;
#endif
}

// drain ASYNCcnt (DScnt handled by __syncthreads) then block barrier
__device__ __forceinline__ void stage_fence() {
#ifdef HAVE_ASYNC_LDS
#  if __has_builtin(__builtin_amdgcn_s_wait_asynccnt)
  __builtin_amdgcn_s_wait_asynccnt(0);
#  else
  asm volatile("s_wait_asynccnt 0" ::: "memory");
#  endif
#endif
  __syncthreads();
}

__device__ __forceinline__ float sigmf(float x) { return 1.f / (1.f + expf(-x)); }

// ---------------------------------------------------------------------------
// Weight packing into per-lane WMMA A-fragment order.
// A fragment (f16 16x32): lane = 16*h + m, half j -> K = 16h + j
// dst index = ((((cot*3+kh)*3+kw)*CB + cb)*32 + lane)*16 + j
// mode 0: OIHW conv weights  w[co][ciOff+ci][kh][kw]
// mode 1: torch ConvTranspose [Cin][Cout][3][3], flipped+IO-swapped:
//         wf[co][ci][kh][kw] = w[ci][co][2-kh][2-kw]
// ---------------------------------------------------------------------------
__global__ void pack_weights(const float* __restrict__ w, _Float16* __restrict__ dst,
                             int CB, int Cout, int CinCnt, int CinOff, int CinSrc,
                             int mode, int total) {
  int idx = blockIdx.x * blockDim.x + threadIdx.x;
  if (idx >= total) return;
  int j    = idx & 15;
  int lane = (idx >> 4) & 31;
  int rem  = idx >> 9;
  int cb   = rem % CB;  rem /= CB;
  int kw   = rem % 3;   rem /= 3;
  int kh   = rem % 3;   rem /= 3;
  int cot  = rem;
  int m = lane & 15, h = lane >> 4;
  int K = 16 * h + j;
  int cil = cb * 32 + K;
  int co  = cot * 16 + m;
  float v = 0.f;
  if (cil < CinCnt && co < Cout) {
    if (mode == 0)
      v = w[(((size_t)co * CinSrc + (CinOff + cil)) * 3 + kh) * 3 + kw];
    else
      v = w[(((size_t)cil * Cout + co) * 3 + (2 - kh)) * 3 + (2 - kw)];
  }
  dst[idx] = (_Float16)v;
}

// Fold conv-bias + eval BatchNorm into per-channel scale/shift
__global__ void bn_prep(const float* __restrict__ bias, const float* __restrict__ g,
                        const float* __restrict__ be, const float* __restrict__ m,
                        const float* __restrict__ v, float* __restrict__ scale,
                        float* __restrict__ shift, int C) {
  int c = blockIdx.x * blockDim.x + threadIdx.x;
  if (c >= C) return;
  float s = g[c] * rsqrtf(v[c] + 1e-5f);
  scale[c] = s;
  shift[c] = (bias[c] - m[c]) * s + be[c];
}

__global__ void log10_kernel(const float* __restrict__ x, float* __restrict__ y, int n) {
  int i = blockIdx.x * blockDim.x + threadIdx.x;
  if (i < n) y[i] = log10f(x[i] + 1e-8f);
}

// enc1: Cin=1 scalar conv 3x3 stride(2,1) + BN + LReLU.
// Writes f32 NCHW (for dec1 skip) and f16 channel-contiguous [b][f][t][c] (for enc2).
__global__ void enc1_kernel(const float* __restrict__ xlog, const float* __restrict__ w,
                            const float* __restrict__ scale, const float* __restrict__ shift,
                            float* __restrict__ e1, _Float16* __restrict__ e1h) {
  int idx = blockIdx.x * blockDim.x + threadIdx.x;      // 16*16*128*512
  int t  = idx & 511;
  int r  = idx >> 9;
  int fo = r & 127; r >>= 7;
  int co = r & 15;
  int b  = r >> 4;
  float acc = 0.f;
#pragma unroll
  for (int kh = 0; kh < 3; ++kh) {
    int fi = 2 * fo + kh - 1;
    if (fi < 0 || fi >= 256) continue;
#pragma unroll
    for (int kw = 0; kw < 3; ++kw) {
      int tt = t + kw - 1;
      if (tt < 0 || tt >= TQ) continue;
      acc += xlog[((size_t)b * 256 + fi) * TQ + tt] * w[co * 9 + kh * 3 + kw];
    }
  }
  float v = acc * scale[co] + shift[co];
  v = (v >= 0.f) ? v : 0.2f * v;
  e1[idx] = v;
  e1h[(((size_t)b * 128 + fo) * TQ + t) * 16 + co] = (_Float16)v;
}

// ---------------------------------------------------------------------------
// Generic WMMA conv, f16 channel-contiguous inputs, implicit channel-concat.
// CB = K-chunks of 32 (compile-time). Grid (T/16, Fout, B), 32*nwaves threads.
// LDS: lds[(kh*18+tt)*CS + ci], CS = 32*CB+8; filled by async global->LDS DMA.
// ---------------------------------------------------------------------------
template <int CB, int KWC>
__global__ void conv_wmma(const _Float16* __restrict__ inAh, int CA,
                          const _Float16* __restrict__ inBh, int CBch,
                          const _Float16* __restrict__ apack,
                          const float* __restrict__ scale, const float* __restrict__ shift,
                          float* __restrict__ outf, _Float16* __restrict__ outh,
                          int Cin, int Cout, int Fin, int Fout, int strideH, int act) {
  constexpr int CINP = 32 * CB;
  constexpr int CS   = CINP + 8;
  extern __shared__ _Float16 lds[];
  const int t0 = blockIdx.x * 16;
  const int fo = blockIdx.y;
  const int b  = blockIdx.z;
  const int tid = threadIdx.x, nthreads = blockDim.x;

  // ---- stage: async copy valid rows, ds-zero invalid rows / channel pad ----
  const int chunksA = CA >> 3, chunks = Cin >> 3;
  const v4f z4 = {};
  for (int idx = tid; idx < 54 * chunks; idx += nthreads) {
    int ck  = idx % chunks;
    int row = idx / chunks;
    int tt  = row % 18;
    int kh  = row / 18;
    int fi  = fo * strideH + kh - 1;
    int t   = t0 - 1 + tt;
    _Float16* ldst = &lds[(kh * 18 + tt) * CS + ck * 8];
    if (fi >= 0 && fi < Fin && t >= 0 && t < TQ) {
      const _Float16* src =
          (ck < chunksA)
              ? inAh + (((size_t)b * Fin + fi) * TQ + t) * CA + ck * 8
              : inBh + (((size_t)b * Fin + fi) * TQ + t) * CBch + (ck - chunksA) * 8;
      a_copy16(src, ldst);
    } else {
      *(v4f*)ldst = z4;
    }
  }
  if (CINP > Cin) {  // zero-pad channels [Cin, CINP)
    int pchunks = (CINP - Cin) >> 3;
    for (int idx = tid; idx < 54 * pchunks; idx += nthreads) {
      int ck = idx % pchunks, row = idx / pchunks;
      *(v4f*)&lds[row * CS + Cin + ck * 8] = z4;
    }
  }
  stage_fence();

  const int lane = tid & 31, wave = tid >> 5, nwaves = nthreads >> 5;
  const int n = lane & 15, hh = lane >> 4;
  const int ntiles = Cout >> 4;
  constexpr int kwb = KWC ? 1 : 0, kwe = KWC ? 2 : 3;

  for (int cot = wave; cot < ntiles; cot += nwaves) {
    v8f acc = {};
#pragma unroll
    for (int kh = 0; kh < 3; ++kh)
#pragma unroll
      for (int kw = kwb; kw < kwe; ++kw)
#pragma unroll
        for (int cb = 0; cb < CB; ++cb) {
          const _Float16* ap =
              apack + ((size_t)((((cot * 3 + kh) * 3 + kw) * CB + cb)) * 32 + lane) * 16;
          v16h a = *(const v16h*)ap;
          v16h bb = lds_v16h(&lds[(kh * 18 + (n + kw)) * CS + cb * 32 + 16 * hh]);
          acc = __builtin_amdgcn_wmma_f32_16x16x32_f16(false, a, false, bb, (short)0, acc,
                                                       false, false);
        }
    float vals[8];
#pragma unroll
    for (int r = 0; r < 8; ++r) {
      int co = cot * 16 + r + 8 * hh;
      float v = acc[r];
      if (scale) v = v * scale[co] + shift[co];
      if (act == 1) v = (v >= 0.f) ? v : 0.2f * v;   // LeakyReLU(0.2)
      else if (act == 2) v = (v >= 0.f) ? v : 0.f;   // ReLU
      vals[r] = v;
    }
    if (outf) {
      float* op = outf + (((size_t)b * Cout + cot * 16 + 8 * hh) * Fout + fo) * TQ + t0 + n;
#pragma unroll
      for (int r = 0; r < 8; ++r) op[(size_t)r * Fout * TQ] = vals[r];
    }
    if (outh) {
      Half8 o;
#pragma unroll
      for (int r = 0; r < 8; ++r) o.h[r] = (_Float16)vals[r];
      *(v4f*)(outh + ((((size_t)b * Fout + fo) * TQ + t0 + n) * Cout + cot * 16 + 8 * hh)) =
          o.f;
    }
  }
}

// ---------------------------------------------------------------------------
// WMMA ConvTranspose2d(k=3, stride=(2,1), pad=1, outpad=(1,0)):
// even fo -> {(fi=fo/2, kh=1)}; odd fo -> {(fi=(fo-1)/2,kh=0), (fi=(fo+1)/2,kh=2)}
// ---------------------------------------------------------------------------
template <int CB>
__global__ void convT_wmma(const _Float16* __restrict__ inAh, int CA,
                           const _Float16* __restrict__ inBh, int CBch,
                           const _Float16* __restrict__ apack,
                           const float* __restrict__ scale, const float* __restrict__ shift,
                           float* __restrict__ outf, _Float16* __restrict__ outh,
                           int Cin, int Cout, int Fin, int act) {
  constexpr int CINP = 32 * CB;
  constexpr int CS   = CINP + 8;
  extern __shared__ _Float16 lds[];
  const int t0 = blockIdx.x * 16;
  const int fo = blockIdx.y;   // 0 .. 2*Fin-1
  const int b  = blockIdx.z;
  const int Fout = 2 * Fin;

  int ntaps, fi_[2], kh_[2];
  if ((fo & 1) == 0) { ntaps = 1; fi_[0] = fo >> 1; kh_[0] = 1; fi_[1] = 0; kh_[1] = 0; }
  else {
    fi_[0] = (fo - 1) >> 1; kh_[0] = 0; ntaps = 1;
    int f1 = (fo + 1) >> 1;
    fi_[1] = f1; kh_[1] = 2;
    if (f1 < Fin) ntaps = 2;
  }

  const int tid = threadIdx.x, nthreads = blockDim.x;
  const int chunksA = CA >> 3, chunks = Cin >> 3;
  const v4f z4 = {};
  for (int idx = tid; idx < 36 * chunks; idx += nthreads) {
    int ck  = idx % chunks;
    int row = idx / chunks;
    int tt  = row % 18;
    int rr  = row / 18;
    int t   = t0 - 1 + tt;
    _Float16* ldst = &lds[(rr * 18 + tt) * CS + ck * 8];
    if (rr < ntaps && t >= 0 && t < TQ) {
      int fi = fi_[rr];
      const _Float16* src =
          (ck < chunksA)
              ? inAh + (((size_t)b * Fin + fi) * TQ + t) * CA + ck * 8
              : inBh + (((size_t)b * Fin + fi) * TQ + t) * CBch + (ck - chunksA) * 8;
      a_copy16(src, ldst);
    } else {
      *(v4f*)ldst = z4;
    }
  }
  if (CINP > Cin) {
    int pchunks = (CINP - Cin) >> 3;
    for (int idx = tid; idx < 36 * pchunks; idx += nthreads) {
      int ck = idx % pchunks, row = idx / pchunks;
      *(v4f*)&lds[row * CS + Cin + ck * 8] = z4;
    }
  }
  stage_fence();

  const int lane = tid & 31, wave = tid >> 5, nwaves = nthreads >> 5;
  const int n = lane & 15, hh = lane >> 4;
  const int ntiles = Cout >> 4;

  for (int cot = wave; cot < ntiles; cot += nwaves) {
    v8f acc = {};
    for (int tap = 0; tap < ntaps; ++tap) {
      int kh = kh_[tap];
#pragma unroll
      for (int kw = 0; kw < 3; ++kw)
#pragma unroll
        for (int cb = 0; cb < CB; ++cb) {
          const _Float16* ap =
              apack + ((size_t)((((cot * 3 + kh) * 3 + kw) * CB + cb)) * 32 + lane) * 16;
          v16h a = *(const v16h*)ap;
          v16h bb = lds_v16h(&lds[(tap * 18 + (n + kw)) * CS + cb * 32 + 16 * hh]);
          acc = __builtin_amdgcn_wmma_f32_16x16x32_f16(false, a, false, bb, (short)0, acc,
                                                       false, false);
        }
    }
    float vals[8];
#pragma unroll
    for (int r = 0; r < 8; ++r) {
      int co = cot * 16 + r + 8 * hh;
      float v = acc[r] * scale[co] + shift[co];
      if (act == 2) v = (v >= 0.f) ? v : 0.f;
      vals[r] = v;
    }
    if (outf) {
      float* op = outf + (((size_t)b * Cout + cot * 16 + 8 * hh) * Fout + fo) * TQ + t0 + n;
#pragma unroll
      for (int r = 0; r < 8; ++r) op[(size_t)r * Fout * TQ] = vals[r];
    }
    if (outh) {
      Half8 o;
#pragma unroll
      for (int r = 0; r < 8; ++r) o.h[r] = (_Float16)vals[r];
      *(v4f*)(outh + ((((size_t)b * Fout + fo) * TQ + t0 + n) * Cout + cot * 16 + 8 * hh)) =
          o.f;
    }
  }
}

// ---------------------------------------------------------------------------
// ConvLSTM recurrence: 32 persistent workgroups, global sense barrier.
// ---------------------------------------------------------------------------
__device__ __forceinline__ void gbar(int* cnt, int* gen, int nblk, int target) {
  __syncthreads();
  if (threadIdx.x == 0) {
    if (__hip_atomic_fetch_add(cnt, 1, __ATOMIC_ACQ_REL, __HIP_MEMORY_SCOPE_AGENT) ==
        nblk - 1) {
      __hip_atomic_store(cnt, 0, __ATOMIC_RELAXED, __HIP_MEMORY_SCOPE_AGENT);
      __hip_atomic_store(gen, target, __ATOMIC_RELEASE, __HIP_MEMORY_SCOPE_AGENT);
    } else {
      while (__hip_atomic_load(gen, __ATOMIC_ACQUIRE, __HIP_MEMORY_SCOPE_AGENT) < target)
        __builtin_amdgcn_s_sleep(1);
    }
  }
  __syncthreads();
}

__global__ void lstm_init(_Float16* __restrict__ hbuf, float* __restrict__ cbuf,
                          int* __restrict__ bar) {
  int i = blockIdx.x * blockDim.x + threadIdx.x;
  if (i < 16384) { hbuf[i] = (_Float16)0.f; cbuf[i] = 0.f; }
  if (i < 2) bar[i] = 0;
}

__global__ void __launch_bounds__(256) lstm_seq(
    const float* __restrict__ gates_x, const float* __restrict__ lstm_b,
    const _Float16* __restrict__ aph, _Float16* __restrict__ hbuf,
    float* __restrict__ cbuf, float* __restrict__ gbuf, _Float16* __restrict__ loh,
    int* __restrict__ bar_cnt, int* __restrict__ bar_gen) {
  const int mt = blockIdx.x;                 // M tile: co0 = mt*16 (32 tiles)
  const int tid = threadIdx.x;
  const int lane = tid & 31, wave = tid >> 5;
  const int nn = wave * 16 + (lane & 15);    // N = b*8+f, 128 total
  const int hh = lane >> 4;
  const int bi = nn >> 3, f0 = nn & 7;
  int gen = 0;
  v4f z4 = {};

  for (int t = 0; t < TQ; ++t) {
    // ---- phase 1: recurrent GEMM tile (M=512, N=128, K=384) ----
    v8f acc;
#pragma unroll
    for (int r = 0; r < 8; ++r) {
      int co = mt * 16 + r + 8 * hh;
      acc[r] = gates_x[(((size_t)bi * 512 + co) * 8 + f0) * TQ + t] + lstm_b[co];
    }
#pragma unroll
    for (int kh = 0; kh < 3; ++kh) {
      int fp = f0 + kh - 1;
      bool ok = (fp >= 0) && (fp < 8);
#pragma unroll
      for (int cb = 0; cb < 4; ++cb) {
        v16h a = *(const v16h*)(aph +
                  ((size_t)(((mt * 3 + kh) * 3 + 1) * 4 + cb) * 32 + lane) * 16);
        HalfVec u;
        u.f4[0] = z4; u.f4[1] = z4;
        if (ok) {
          const _Float16* hp = hbuf + (size_t)(bi * 8 + fp) * 128 + cb * 32 + 16 * hh;
          u.f4[0] = *(const v4f*)hp;
          u.f4[1] = *(const v4f*)(hp + 8);
        }
        acc = __builtin_amdgcn_wmma_f32_16x16x32_f16(false, a, false, u.v, (short)0, acc,
                                                     false, false);
      }
    }
#pragma unroll
    for (int r = 0; r < 8; ++r) {
      int co = mt * 16 + r + 8 * hh;
      gbuf[co * 128 + nn] = acc[r];
    }
    __threadfence();
    gbar(bar_cnt, bar_gen, 32, ++gen);

    // ---- phase 2: gate update (128 ch x 128 n over 8192 threads) ----
    for (int e = blockIdx.x * blockDim.x + tid; e < 128 * 128; e += 32 * 256) {
      int ch = e >> 7, n2 = e & 127;
      float ig = gbuf[ch * 128 + n2];
      float fg = gbuf[(128 + ch) * 128 + n2];
      float og = gbuf[(256 + ch) * 128 + n2];
      float gg = gbuf[(384 + ch) * 128 + n2];
      float cnew = sigmf(fg) * cbuf[e] + sigmf(ig) * tanhf(gg);
      float hval = sigmf(og) * tanhf(cnew);
      cbuf[e] = cnew;
      int b2 = n2 >> 3, f2 = n2 & 7;
      hbuf[(b2 * 8 + f2) * 128 + ch] = (_Float16)hval;
      loh[(((size_t)(b2 * 8 + f2)) * TQ + t) * 128 + ch] = (_Float16)hval;
    }
    __threadfence();
    gbar(bar_cnt, bar_gen, 32, ++gen);
  }
}

// dec1: Cout=1 transposed conv on concat(d2,e1) + sigmoid mask + multiply by x
__global__ void dec1_kernel(const float* __restrict__ d2, const float* __restrict__ e1,
                            const float* __restrict__ w, const float* __restrict__ bias,
                            const float* __restrict__ x, float* __restrict__ out) {
  int idx = blockIdx.x * blockDim.x + threadIdx.x;   // 16*256*512
  int t  = idx & 511;
  int r  = idx >> 9;
  int fo = r & 255;
  int b  = r >> 8;

  int ntaps, fi_[2], kh_[2];
  if ((fo & 1) == 0) { ntaps = 1; fi_[0] = fo >> 1; kh_[0] = 1; fi_[1] = 0; kh_[1] = 0; }
  else {
    fi_[0] = (fo - 1) >> 1; kh_[0] = 0; ntaps = 1;
    int f1 = (fo + 1) >> 1;
    fi_[1] = f1; kh_[1] = 2;
    if (f1 < 128) ntaps = 2;
  }
  float acc = bias[0];
  for (int tap = 0; tap < ntaps; ++tap) {
    int fi = fi_[tap], kh = kh_[tap];
    for (int kw = 0; kw < 3; ++kw) {
      int t2 = t + kw - 1;
      if (t2 < 0 || t2 >= TQ) continue;
      for (int ci = 0; ci < 32; ++ci) {
        float xv = (ci < 16) ? d2[(((size_t)b * 16 + ci) * 128 + fi) * TQ + t2]
                             : e1[(((size_t)b * 16 + (ci - 16)) * 128 + fi) * TQ + t2];
        acc += xv * w[ci * 9 + (2 - kh) * 3 + (2 - kw)];
      }
    }
  }
  out[idx] = x[idx] * sigmf(acc);
}

// ---------------------------------------------------------------------------
extern "C" void kernel_launch(void* const* d_in, const int* in_sizes, int n_in,
                              void* d_out, int out_size, void* d_ws, size_t ws_size,
                              hipStream_t stream) {
  (void)in_sizes; (void)n_in; (void)out_size;
  const float* x = (const float*)d_in[0];
  const float *w_e[5], *b_e[5], *g_e[5], *be_e[5], *m_e[5], *v_e[5];
  for (int i = 0; i < 5; ++i) {
    w_e[i]  = (const float*)d_in[1 + 6 * i + 0];
    b_e[i]  = (const float*)d_in[1 + 6 * i + 1];
    g_e[i]  = (const float*)d_in[1 + 6 * i + 2];
    be_e[i] = (const float*)d_in[1 + 6 * i + 3];
    m_e[i]  = (const float*)d_in[1 + 6 * i + 4];
    v_e[i]  = (const float*)d_in[1 + 6 * i + 5];
  }
  const float* lstm_w = (const float*)d_in[31];
  const float* lstm_b = (const float*)d_in[32];
  const float *w_d[4], *b_d[4], *g_d[4], *be_d[4], *m_d[4], *v_d[4];  // dec5..dec2
  for (int k = 0; k < 4; ++k) {
    w_d[k]  = (const float*)d_in[33 + 6 * k + 0];
    b_d[k]  = (const float*)d_in[33 + 6 * k + 1];
    g_d[k]  = (const float*)d_in[33 + 6 * k + 2];
    be_d[k] = (const float*)d_in[33 + 6 * k + 3];
    m_d[k]  = (const float*)d_in[33 + 6 * k + 4];
    v_d[k]  = (const float*)d_in[33 + 6 * k + 5];
  }
  const float* dec1_w = (const float*)d_in[57];
  const float* dec1_b = (const float*)d_in[58];

  char* ws = (char*)d_ws;
  size_t off = 0;
  auto alloc = [&](size_t bytes) -> char* {
    off = (off + 255) & ~(size_t)255;
    char* p = ws + off;
    off += bytes;
    return p;
  };

  const size_t NACT = 16777216ull;
  // f32 buffers (only where scalar consumers need them)
  float* xlog  = (float*)alloc(2097152ull * 4);
  float* e1    = (float*)alloc(NACT * 4);
  float* gates = (float*)alloc(33554432ull * 4);  // region reused for d4h/d3h after LSTM
  float* d2f   = (float*)alloc(NACT * 4);
  // f16 channel-contiguous activations [b][f][t][c]
  _Float16* e1h = (_Float16*)alloc(NACT * 2);
  _Float16* e2h = (_Float16*)alloc(NACT * 2);
  _Float16* e3h = (_Float16*)alloc(NACT * 2);
  _Float16* e4h = (_Float16*)alloc(NACT * 2);
  _Float16* e5h = (_Float16*)alloc(NACT * 2);
  _Float16* d5h = e5h;                               // e5h dead after gates conv
  _Float16* loh = (_Float16*)alloc(8388608ull * 2);  // lstm output f16T
  _Float16* d4h = (_Float16*)gates;                  // gates dead after recurrence
  _Float16* d3h = (_Float16*)((char*)gates + NACT * 2);
  // LSTM state
  _Float16* hbuf = (_Float16*)alloc(16384 * 2);
  float* cbuf = (float*)alloc(16384 * 4);
  float* gbuf = (float*)alloc(65536 * 4);
  int* bar = (int*)alloc(256);
  // packed weights (halves): ntiles*9*CB*512
  _Float16* pk_e2 = (_Float16*)alloc((size_t)2 * 9 * 1 * 512 * 2);
  _Float16* pk_e3 = (_Float16*)alloc((size_t)4 * 9 * 1 * 512 * 2);
  _Float16* pk_e4 = (_Float16*)alloc((size_t)8 * 9 * 2 * 512 * 2);
  _Float16* pk_e5 = (_Float16*)alloc((size_t)16 * 9 * 4 * 512 * 2);
  _Float16* pk_lx = (_Float16*)alloc((size_t)32 * 9 * 8 * 512 * 2);
  _Float16* pk_lh = (_Float16*)alloc((size_t)32 * 9 * 4 * 512 * 2);
  _Float16* pk_d5 = (_Float16*)alloc((size_t)8 * 9 * 4 * 512 * 2);
  _Float16* pk_d4 = (_Float16*)alloc((size_t)4 * 9 * 8 * 512 * 2);
  _Float16* pk_d3 = (_Float16*)alloc((size_t)2 * 9 * 4 * 512 * 2);
  _Float16* pk_d2 = (_Float16*)alloc((size_t)1 * 9 * 2 * 512 * 2);

  const int coutE[5] = {16, 32, 64, 128, 256};
  const int coutD[4] = {128, 64, 32, 16};
  float *scE[5], *shE[5], *scD[4], *shD[4];
  for (int i = 0; i < 5; ++i) { scE[i] = (float*)alloc(coutE[i] * 4); shE[i] = (float*)alloc(coutE[i] * 4); }
  for (int k = 0; k < 4; ++k) { scD[k] = (float*)alloc(coutD[k] * 4); shD[k] = (float*)alloc(coutD[k] * 4); }

  if (off > ws_size) return;

  // ---- BN fold + weight packing ----
  for (int i = 0; i < 5; ++i)
    bn_prep<<<(coutE[i] + 63) / 64, 64, 0, stream>>>(b_e[i], g_e[i], be_e[i], m_e[i], v_e[i],
                                                     scE[i], shE[i], coutE[i]);
  for (int k = 0; k < 4; ++k)
    bn_prep<<<(coutD[k] + 63) / 64, 64, 0, stream>>>(b_d[k], g_d[k], be_d[k], m_d[k], v_d[k],
                                                     scD[k], shD[k], coutD[k]);

  auto pack = [&](const float* w, _Float16* dst, int ntiles, int CB, int Cout, int CinCnt,
                  int CinOff, int CinSrc, int mode) {
    int total = ntiles * 9 * CB * 512;
    pack_weights<<<(total + 255) / 256, 256, 0, stream>>>(w, dst, CB, Cout, CinCnt, CinOff,
                                                          CinSrc, mode, total);
  };
  pack(w_e[1], pk_e2, 2, 1, 32, 16, 0, 16, 0);
  pack(w_e[2], pk_e3, 4, 1, 64, 32, 0, 32, 0);
  pack(w_e[3], pk_e4, 8, 2, 128, 64, 0, 64, 0);
  pack(w_e[4], pk_e5, 16, 4, 256, 128, 0, 128, 0);
  pack(lstm_w, pk_lx, 32, 8, 512, 256, 0, 384, 0);
  pack(lstm_w, pk_lh, 32, 4, 512, 128, 256, 384, 0);
  pack(w_d[0], pk_d5, 8, 4, 128, 128, 0, 128, 1);
  pack(w_d[1], pk_d4, 4, 8, 64, 256, 0, 256, 1);
  pack(w_d[2], pk_d3, 2, 4, 32, 128, 0, 128, 1);
  pack(w_d[3], pk_d2, 1, 2, 16, 64, 0, 64, 1);

  // ---- forward pass ----
  log10_kernel<<<8192, 256, 0, stream>>>(x, xlog, 2097152);
  enc1_kernel<<<65536, 256, 0, stream>>>(xlog, w_e[0], scE[0], shE[0], e1, e1h);

  // encoders (f16T in -> f16T out)
  conv_wmma<1, 0><<<dim3(32, 64, BQ), 64, (size_t)3 * 18 * 40 * 2, stream>>>(
      e1h, 16, nullptr, 0, pk_e2, scE[1], shE[1], nullptr, e2h, 16, 32, 128, 64, 2, 1);
  conv_wmma<1, 0><<<dim3(32, 32, BQ), 128, (size_t)3 * 18 * 40 * 2, stream>>>(
      e2h, 32, nullptr, 0, pk_e3, scE[2], shE[2], nullptr, e3h, 32, 64, 64, 32, 2, 1);
  conv_wmma<2, 0><<<dim3(32, 16, BQ), 256, (size_t)3 * 18 * 72 * 2, stream>>>(
      e3h, 64, nullptr, 0, pk_e4, scE[3], shE[3], nullptr, e4h, 64, 128, 32, 16, 2, 1);
  conv_wmma<4, 0><<<dim3(32, 8, BQ), 256, (size_t)3 * 18 * 136 * 2, stream>>>(
      e4h, 128, nullptr, 0, pk_e5, scE[4], shE[4], nullptr, e5h, 128, 256, 16, 8, 2, 1);

  // LSTM input projection (3x1 conv over F, center kw column), f32 out for recurrence
  conv_wmma<8, 1><<<dim3(32, 8, BQ), 256, (size_t)3 * 18 * 264 * 2, stream>>>(
      e5h, 256, nullptr, 0, pk_lx, nullptr, nullptr, gates, nullptr, 256, 512, 8, 8, 1, 0);

  // LSTM recurrence (32 persistent, co-resident workgroups + global barrier)
  lstm_init<<<64, 256, 0, stream>>>(hbuf, cbuf, bar);
  lstm_seq<<<32, 256, 0, stream>>>(gates, lstm_b, pk_lh, hbuf, cbuf, gbuf, loh,
                                   bar, bar + 1);

  // decoders (transposed convs; channel-concat via dual async-staged sources)
  convT_wmma<4><<<dim3(32, 16, BQ), 256, (size_t)2 * 18 * 136 * 2, stream>>>(
      loh, 128, nullptr, 0, pk_d5, scD[0], shD[0], nullptr, d5h, 128, 128, 8, 2);
  convT_wmma<8><<<dim3(32, 32, BQ), 128, (size_t)2 * 18 * 264 * 2, stream>>>(
      d5h, 128, e4h, 128, pk_d4, scD[1], shD[1], nullptr, d4h, 256, 64, 16, 2);
  convT_wmma<4><<<dim3(32, 64, BQ), 64, (size_t)2 * 18 * 136 * 2, stream>>>(
      d4h, 64, e3h, 64, pk_d3, scD[2], shD[2], nullptr, d3h, 128, 32, 32, 2);
  convT_wmma<2><<<dim3(32, 128, BQ), 32, (size_t)2 * 18 * 72 * 2, stream>>>(
      d3h, 32, e2h, 32, pk_d2, scD[3], shD[3], d2f, nullptr, 64, 16, 64, 2);

  dec1_kernel<<<8192, 256, 0, stream>>>(d2f, e1, dec1_w, dec1_b, x, (float*)d_out);
}
